// RWKV_TimeMix_v7_65163243815256
// MI455X (gfx1250) — matrix operand order repaired and checked by reading, not compiled, not run on hardware
//
#include <hip/hip_runtime.h>
#include <hip/hip_bf16.h>
#include <math.h>

// ---------------- problem constants ----------------
#define B_   8
#define T_   2048
#define C_   1024
#define H_   16
#define D_   64
#define BT_  (B_*T_)          // 16384 token rows
#define SEG_ 16               // scan segments
#define SEGLEN_ (T_/SEG_)     // 128

typedef __attribute__((ext_vector_type(16))) __bf16 v16bf;
typedef __attribute__((ext_vector_type(8)))  float  v8f;

__device__ __forceinline__ __bf16 f2bf(float f) {
  unsigned u = __builtin_bit_cast(unsigned, f);
  unsigned short h = (unsigned short)((u + 0x7FFFu + ((u >> 16) & 1u)) >> 16);
  return __builtin_bit_cast(__bf16, h);
}

__device__ __forceinline__ float sigm(float x) { return 1.f / (1.f + expf(-x)); }

// low 32 bits of an LDS flat address are the LDS byte offset (ISA 10.2 aperture rule)
__device__ __forceinline__ unsigned lds_off(const void* p) {
  return (unsigned)(unsigned long long)(uintptr_t)p;
}

// async DMA of 16 bytes global -> LDS, tracked by ASYNCcnt (CDNA5 ISA)
__device__ __forceinline__ void async_ld_b128(unsigned ldsaddr, unsigned long long gaddr) {
  asm volatile("global_load_async_to_lds_b128 %0, %1, off"
               :: "v"(ldsaddr), "v"(gaddr) : "memory");
}
__device__ __forceinline__ void wait_async0() {
  asm volatile("s_wait_asynccnt 0x0" ::: "memory");
}

// block-wide (64 thread) sum reduction via LDS
__device__ __forceinline__ float bsum64(float* buf, int tid, float v) {
  buf[tid] = v; __syncthreads();
  if (tid < 32) buf[tid] += buf[tid + 32]; __syncthreads();
  if (tid < 16) buf[tid] += buf[tid + 16]; __syncthreads();
  if (tid <  8) buf[tid] += buf[tid +  8]; __syncthreads();
  if (tid <  4) buf[tid] += buf[tid +  4]; __syncthreads();
  if (tid <  2) buf[tid] += buf[tid +  2]; __syncthreads();
  if (tid <  1) buf[tid] += buf[tid +  1]; __syncthreads();
  float r = buf[0]; __syncthreads();
  return r;
}

// ---------------- weight transpose fp32 -> bf16 : dst[c*R + r] = src[r*Cc + c]
__global__ void k_transpose_bf16(const float* __restrict__ src, __bf16* __restrict__ dst,
                                 int R, int Cc) {
  __shared__ float tile[32][33];
  int c0 = blockIdx.x * 32, r0 = blockIdx.y * 32;
  for (int yy = 0; yy < 32; yy += 8)
    tile[threadIdx.y + yy][threadIdx.x] =
        src[(size_t)(r0 + threadIdx.y + yy) * Cc + c0 + threadIdx.x];
  __syncthreads();
  for (int yy = 0; yy < 32; yy += 8)
    dst[(size_t)(c0 + threadIdx.y + yy) * R + r0 + threadIdx.x] =
        f2bf(tile[threadIdx.x][threadIdx.y + yy]);
}

// ---------------- token shift + 6 mixes, fp32 -> bf16
__global__ void k_mix(const float* __restrict__ x,
                      const float* __restrict__ mr, const float* __restrict__ mw,
                      const float* __restrict__ mk, const float* __restrict__ mv,
                      const float* __restrict__ ma, const float* __restrict__ mg,
                      __bf16* __restrict__ xr, __bf16* __restrict__ xw,
                      __bf16* __restrict__ xk, __bf16* __restrict__ xv,
                      __bf16* __restrict__ xa, __bf16* __restrict__ xg) {
  int e = blockIdx.x * 256 + threadIdx.x;
  if (e >= BT_ * C_) return;
  int c = e & (C_ - 1);
  int t = (e >> 10) & (T_ - 1);
  float xc = x[e];
  float xp = (t == 0) ? 0.f : x[e - C_];
  float sx = xc - xp;
  xr[e] = f2bf(xc + sx * mr[c]);
  xw[e] = f2bf(xc + sx * mw[c]);
  xk[e] = f2bf(xc + sx * mk[c]);
  xv[e] = f2bf(xc + sx * mv[c]);
  xa[e] = f2bf(xc + sx * ma[c]);
  xg[e] = f2bf(xc + sx * mg[c]);
}

// ---------------- bf16 WMMA GEMM: C[M,N] = A[M,K] * Bt[N,K]^T ----------------
// 8 waves/block arranged WM x WN, each wave computes a 32x32 quad of 16x16 tiles.
// Ping-pong LDS buffers filled by GLOBAL_LOAD_ASYNC_TO_LDS_B128 (ASYNCcnt),
// so the next K-tile's DMA overlaps the current tile's WMMAs.
// LDS rows padded to 40 bf16 (80B) -> conflict-free ds_load_b128 fragment loads.
template <int BM, int BN, int WM, int WN>
__launch_bounds__(256)
__global__ void k_gemm_bf16(const __bf16* __restrict__ A, const __bf16* __restrict__ Bt,
                            float* __restrict__ Cout, int M, int N, int K) {
  constexpr int LDT = 40;
  constexpr int ACH = BM / 64;   // 16B chunks per thread for A tile
  constexpr int BCH = BN / 64;   // 16B chunks per thread for B tile
  __shared__ __attribute__((aligned(16))) __bf16 lA[2][BM * LDT];
  __shared__ __attribute__((aligned(16))) __bf16 lB[2][BN * LDT];
  int tid  = threadIdx.x;
  int lane = tid & 31, wave = tid >> 5;
  int hi = lane >> 4, lm = lane & 15;
  int wm = wave % WM, wn = wave / WM;
  int mBase = blockIdx.x * BM;
  int nBase = blockIdx.y * BN;

  // per-thread staging addresses (advance 64B per K-tile)
  unsigned long long gA[ACH], gB[BCH];
  unsigned loA[2][ACH], loB[2][BCH];
#pragma unroll
  for (int ii = 0; ii < ACH; ii++) {
    int ch = tid + ii * 256, r = ch >> 2, cc = ch & 3;
    gA[ii] = (unsigned long long)(uintptr_t)(A + (size_t)(mBase + r) * K + cc * 8);
    loA[0][ii] = lds_off(&lA[0][r * LDT + cc * 8]);
    loA[1][ii] = lds_off(&lA[1][r * LDT + cc * 8]);
  }
#pragma unroll
  for (int ii = 0; ii < BCH; ii++) {
    int ch = tid + ii * 256, r = ch >> 2, cc = ch & 3;
    gB[ii] = (unsigned long long)(uintptr_t)(Bt + (size_t)(nBase + r) * K + cc * 8);
    loB[0][ii] = lds_off(&lB[0][r * LDT + cc * 8]);
    loB[1][ii] = lds_off(&lB[1][r * LDT + cc * 8]);
  }

  auto issue_tile = [&](int buf) {
#pragma unroll
    for (int ii = 0; ii < ACH; ii++) { async_ld_b128(loA[buf][ii], gA[ii]); gA[ii] += 64; }
#pragma unroll
    for (int ii = 0; ii < BCH; ii++) { async_ld_b128(loB[buf][ii], gB[ii]); gB[ii] += 64; }
  };

  v8f acc[2][2] = {};

  issue_tile(0);
  wait_async0();
  __syncthreads();

  int nIter = K >> 5;
  for (int it = 0; it < nIter; it++) {
    int cur = it & 1;
    if (it + 1 < nIter) issue_tile(cur ^ 1);   // DMA next tile during compute

    v16bf afrag[2], bfrag[2];
    // A fragment: lo lanes K0-7 & K16-23, hi lanes K8-15 & K24-31
#pragma unroll
    for (int ms = 0; ms < 2; ms++) {
      const __bf16* p = &lA[cur][(wm * 32 + ms * 16 + lm) * LDT];
      *((float4*)&afrag[ms])     = *(const float4*)(p + hi * 8);
      *((float4*)&afrag[ms] + 1) = *(const float4*)(p + 16 + hi * 8);
    }
    // B fragment: lo lanes K0-7 & K16-23, hi lanes K8-15 & K24-31 (lane = N column)
#pragma unroll
    for (int ns = 0; ns < 2; ns++) {
      const __bf16* p = &lB[cur][(wn * 32 + ns * 16 + lm) * LDT];
      *((float4*)&bfrag[ns])     = *(const float4*)(p + hi * 8);
      *((float4*)&bfrag[ns] + 1) = *(const float4*)(p + 16 + hi * 8);
    }
#pragma unroll
    for (int ms = 0; ms < 2; ms++)
#pragma unroll
      for (int ns = 0; ns < 2; ns++)
        acc[ms][ns] = __builtin_amdgcn_wmma_f32_16x16x32_bf16(
            false, afrag[ms], false, bfrag[ns], (short)0, acc[ms][ns], false, false);

    wait_async0();      // next buffer fully landed in LDS
    __syncthreads();    // + all waves done reading cur (safe to overwrite next iter)
  }

  // store: VGPR j of D tile -> row (hi*8 + j), col lm
#pragma unroll
  for (int ms = 0; ms < 2; ms++)
#pragma unroll
    for (int ns = 0; ns < 2; ns++) {
      int rb = mBase + wm * 32 + ms * 16 + hi * 8;
      int cb = nBase + wn * 32 + ns * 16 + lm;
#pragma unroll
      for (int j = 0; j < 8; j++)
        Cout[(size_t)(rb + j) * N + cb] = acc[ms][ns][j];
    }
}

// ---------------- LoRA stage 2 for decay (w) and in-context lr (a) ----------
// one wave per token; lanes 0-15 -> w head, lanes 16-31 -> a head
__global__ void k_stage2_wa(const float* __restrict__ hw, const float* __restrict__ ha,
                            const float* __restrict__ w0, const float* __restrict__ w2,
                            const float* __restrict__ a0, const float* __restrict__ a2,
                            float* __restrict__ wdec, float* __restrict__ acoef) {
  int tok = blockIdx.x;
  int lane = threadIdx.x;
  int h = lane & 15;
  bool isW = lane < 16;
  const float* hsrc = isW ? hw : ha;
  const float* m2   = isW ? w2 : a2;
  float z = isW ? w0[h] : a0[h];
  for (int j = 0; j < 64; j++)
    z += tanhf(hsrc[tok * 64 + j]) * m2[j * 16 + h];
  if (isW) {
    float y  = -z;
    float sp = (y > 20.f) ? y : log1pf(expf(y));   // softplus(-z)
    wdec[tok * 16 + h] = expf(-sp - 0.5f);
  } else {
    acoef[tok * 16 + h] = sigm(z);
  }
}

__global__ void k_tanh_bf16(const float* __restrict__ in, __bf16* __restrict__ outp, int n) {
  int i = blockIdx.x * 256 + threadIdx.x;
  if (i < n) outp[i] = f2bf(tanhf(in[i]));
}

// ---------------- kk normalize (in kkbuf) + kv = kk*v*a (in-place over k) ----
__global__ void k_prep(float* __restrict__ kbuf, float* __restrict__ kkbuf,
                       const float* __restrict__ vbuf, const float* __restrict__ k_k,
                       const float* __restrict__ acoef) {
  __shared__ float buf[64];
  int bidx = blockIdx.x;            // tok*H + h
  int h = bidx & 15;
  int d = threadIdx.x;
  int idx = bidx * 64 + d;
  float kx = kbuf[idx] * k_k[h * 64 + d];
  float ss = bsum64(buf, d, kx * kx);
  float kkv = kx / fmaxf(sqrtf(ss), 1e-12f);
  kkbuf[idx] = kkv;
  kbuf[idx]  = kkv * vbuf[idx] * acoef[bidx];   // kv
}

// ---------------- 3-pass chunked scan over T ----------------
// pass1: per-(b,h,seg) local scan + per-step cumulative decay product
__global__ void k_scan1(const float* __restrict__ kv, const float* __restrict__ wdec,
                        float* __restrict__ states, float* __restrict__ cumw) {
  int blk = blockIdx.x;             // ((b*H)+h)*SEG + s
  int s = blk & (SEG_ - 1);
  int h = (blk >> 4) & (H_ - 1);
  int b = blk >> 8;
  int d = threadIdx.x;
  float st = 0.f, cp = 1.f;
  int t0 = s * SEGLEN_;
  for (int i = 0; i < SEGLEN_; i++) {
    int th = (b * T_ + t0 + i) * H_ + h;
    float w = wdec[th];
    cp *= w;
    st = st * w + kv[(size_t)th * 64 + d];
    states[(size_t)th * 64 + d] = st;
    if (d == 0) cumw[th] = cp;
  }
}

// pass2: sequential carry across the 16 segments, per (b,h,d)
__global__ void k_scan2(const float* __restrict__ states, const float* __restrict__ cumw,
                        float* __restrict__ carry) {
  int blk = blockIdx.x;             // b*H + h
  int h = blk & 15, b = blk >> 4;
  int d = threadIdx.x;
  float c = 0.f;
  for (int s = 0; s < SEG_; s++) {
    carry[(size_t)((b * SEG_ + s) * H_ + h) * 64 + d] = c;
    int tlast = s * SEGLEN_ + SEGLEN_ - 1;
    int th = (b * T_ + tlast) * H_ + h;
    c = c * cumw[th] + states[(size_t)th * 64 + d];
  }
}

// pass3: states += carry(seg) * cumprod(w within seg up to t)
__global__ void k_scan3(float* __restrict__ states, const float* __restrict__ carry,
                        const float* __restrict__ cumw) {
  int i = blockIdx.x * 256 + threadIdx.x;
  if (i >= BT_ * C_) return;
  int th = i >> 6;                  // (b*T+t)*H + h
  int d = i & 63;
  int h = th & 15;
  int t = (th >> 4) & (T_ - 1);
  int b = th >> 15;                 // T_*H_ = 2^15
  int s = t >> 7;                   // SEGLEN_ = 128
  states[i] += carry[(size_t)((b * SEG_ + s) * H_ + h) * 64 + d] * cumw[th];
}

// ---------------- bonus + GroupNorm + gate, produce bf16 input of Wo --------
__global__ void k_post(const float* __restrict__ states, const float* __restrict__ r,
                       const float* __restrict__ kk, const float* __restrict__ v,
                       const float* __restrict__ r_k, const float* __restrict__ glin,
                       const float* __restrict__ ln_w, const float* __restrict__ ln_b,
                       __bf16* __restrict__ xng) {
  __shared__ float buf[64];
  int bidx = blockIdx.x;            // tok*H + h
  int h = bidx & 15;
  int d = threadIdx.x;
  int idx = bidx * 64 + d;
  float rv = r[idx], kkv = kk[idx], vh = v[idx], st = states[idx];
  float bonus = bsum64(buf, d, rv * kkv * r_k[h * 64 + d]);
  float xw = st * rv + bonus * vh;
  float mean = bsum64(buf, d, xw) * (1.f / 64.f);
  float var  = bsum64(buf, d, xw * xw) * (1.f / 64.f) - mean * mean;
  float xn = (xw - mean) * rsqrtf(var + 1e-5f);
  int c = h * 64 + d;
  float y = xn * ln_w[c] + ln_b[c];
  y *= sigm(glin[idx]);
  xng[idx] = f2bf(y);
}

// ============================ host launch ============================
extern "C" void kernel_launch(void* const* d_in, const int* in_sizes, int n_in,
                              void* d_out, int out_size, void* d_ws, size_t ws_size,
                              hipStream_t stream) {
  (void)in_sizes; (void)n_in; (void)out_size; (void)ws_size;
  const float* x    = (const float*)d_in[0];
  const float* x_r  = (const float*)d_in[1];
  const float* x_w  = (const float*)d_in[2];
  const float* x_k  = (const float*)d_in[3];
  const float* x_v  = (const float*)d_in[4];
  const float* x_a  = (const float*)d_in[5];
  const float* x_g  = (const float*)d_in[6];
  const float* w0   = (const float*)d_in[7];
  const float* w1   = (const float*)d_in[8];
  const float* w2   = (const float*)d_in[9];
  const float* a0   = (const float*)d_in[10];
  const float* a1   = (const float*)d_in[11];
  const float* a2   = (const float*)d_in[12];
  // d_in[13..15] = v0,v1,v2 : value-residual branch is a mathematical no-op
  const float* g1   = (const float*)d_in[16];
  const float* g2   = (const float*)d_in[17];
  const float* k_k  = (const float*)d_in[18];
  const float* r_k  = (const float*)d_in[19];
  const float* Wr   = (const float*)d_in[20];
  const float* Wk   = (const float*)d_in[21];
  const float* Wv   = (const float*)d_in[22];
  const float* Wo   = (const float*)d_in[23];
  const float* ln_w = (const float*)d_in[24];
  const float* ln_b = (const float*)d_in[25];
  float* out = (float*)d_out;

  // workspace carve-up (~635 MB total)
  char* ws = (char*)d_ws;
  size_t off = 0;
  auto alloc = [&](size_t bytes) -> char* {
    char* p = ws + off;
    off = (off + bytes + 255) & ~(size_t)255;
    return p;
  };
  __bf16* WrT  = (__bf16*)alloc((size_t)C_ * C_ * 2);
  __bf16* WkT  = (__bf16*)alloc((size_t)C_ * C_ * 2);
  __bf16* WvT  = (__bf16*)alloc((size_t)C_ * C_ * 2);
  __bf16* WoT  = (__bf16*)alloc((size_t)C_ * C_ * 2);
  __bf16* w1T  = (__bf16*)alloc((size_t)64 * C_ * 2);
  __bf16* a1T  = (__bf16*)alloc((size_t)64 * C_ * 2);
  __bf16* g1T  = (__bf16*)alloc((size_t)64 * C_ * 2);
  __bf16* g2T  = (__bf16*)alloc((size_t)C_ * 64 * 2);
  __bf16* xr   = (__bf16*)alloc((size_t)BT_ * C_ * 2);
  __bf16* xw   = (__bf16*)alloc((size_t)BT_ * C_ * 2);
  __bf16* xk   = (__bf16*)alloc((size_t)BT_ * C_ * 2);
  __bf16* xv   = (__bf16*)alloc((size_t)BT_ * C_ * 2);
  __bf16* xa   = (__bf16*)alloc((size_t)BT_ * C_ * 2);
  __bf16* xg   = (__bf16*)alloc((size_t)BT_ * C_ * 2);
  float* rbuf  = (float*)alloc((size_t)BT_ * C_ * 4);
  float* kbuf  = (float*)alloc((size_t)BT_ * C_ * 4);   // k, then kv in-place
  float* vbuf  = (float*)alloc((size_t)BT_ * C_ * 4);
  float* kkbuf = (float*)alloc((size_t)BT_ * C_ * 4);
  float* hw    = (float*)alloc((size_t)BT_ * 64 * 4);
  float* ha    = (float*)alloc((size_t)BT_ * 64 * 4);
  float* hg    = (float*)alloc((size_t)BT_ * 64 * 4);
  __bf16* thg  = (__bf16*)alloc((size_t)BT_ * 64 * 2);
  float* glin  = (float*)alloc((size_t)BT_ * C_ * 4);
  float* wdec  = (float*)alloc((size_t)BT_ * H_ * 4);
  float* acoef = (float*)alloc((size_t)BT_ * H_ * 4);
  float* cumw  = (float*)alloc((size_t)BT_ * H_ * 4);
  float* carry = (float*)alloc((size_t)B_ * SEG_ * H_ * D_ * 4);
  float* states= (float*)alloc((size_t)BT_ * C_ * 4);
  __bf16* xng  = (__bf16*)alloc((size_t)BT_ * C_ * 2);

  dim3 tb(32, 8);
  // weight transposes (fp32 [K,N] -> bf16 [N,K])
  k_transpose_bf16<<<dim3(C_ / 32, C_ / 32), tb, 0, stream>>>(Wr, WrT, C_, C_);
  k_transpose_bf16<<<dim3(C_ / 32, C_ / 32), tb, 0, stream>>>(Wk, WkT, C_, C_);
  k_transpose_bf16<<<dim3(C_ / 32, C_ / 32), tb, 0, stream>>>(Wv, WvT, C_, C_);
  k_transpose_bf16<<<dim3(C_ / 32, C_ / 32), tb, 0, stream>>>(Wo, WoT, C_, C_);
  k_transpose_bf16<<<dim3(64 / 32, C_ / 32), tb, 0, stream>>>(w1, w1T, C_, 64);
  k_transpose_bf16<<<dim3(64 / 32, C_ / 32), tb, 0, stream>>>(a1, a1T, C_, 64);
  k_transpose_bf16<<<dim3(64 / 32, C_ / 32), tb, 0, stream>>>(g1, g1T, C_, 64);
  k_transpose_bf16<<<dim3(C_ / 32, 64 / 32), tb, 0, stream>>>(g2, g2T, 64, C_);

  int nel = BT_ * C_;
  k_mix<<<(nel + 255) / 256, 256, 0, stream>>>(x, x_r, x_w, x_k, x_v, x_a, x_g,
                                               xr, xw, xk, xv, xa, xg);

  dim3 gBig(BT_ / 64, C_ / 128);    // 256 x 8 blocks
  k_gemm_bf16<64, 128, 2, 4><<<gBig, 256, 0, stream>>>(xr, WrT, rbuf, BT_, C_, C_);
  k_gemm_bf16<64, 128, 2, 4><<<gBig, 256, 0, stream>>>(xk, WkT, kbuf, BT_, C_, C_);
  k_gemm_bf16<64, 128, 2, 4><<<gBig, 256, 0, stream>>>(xv, WvT, vbuf, BT_, C_, C_);

  dim3 gLora(BT_ / 128, 1);
  k_gemm_bf16<128, 64, 4, 2><<<gLora, 256, 0, stream>>>(xw, w1T, hw, BT_, 64, C_);
  k_gemm_bf16<128, 64, 4, 2><<<gLora, 256, 0, stream>>>(xa, a1T, ha, BT_, 64, C_);
  k_gemm_bf16<128, 64, 4, 2><<<gLora, 256, 0, stream>>>(xg, g1T, hg, BT_, 64, C_);

  k_stage2_wa<<<BT_, 32, 0, stream>>>(hw, ha, w0, w2, a0, a2, wdec, acoef);
  k_tanh_bf16<<<(BT_ * 64 + 255) / 256, 256, 0, stream>>>(hg, thg, BT_ * 64);
  k_gemm_bf16<64, 128, 2, 4><<<gBig, 256, 0, stream>>>(thg, g2T, glin, BT_, C_, 64);

  k_prep<<<BT_ * H_, 64, 0, stream>>>(kbuf, kkbuf, vbuf, k_k, acoef);

  k_scan1<<<B_ * H_ * SEG_, 64, 0, stream>>>(kbuf, wdec, states, cumw);
  k_scan2<<<B_ * H_, 64, 0, stream>>>(states, cumw, carry);
  k_scan3<<<(nel + 255) / 256, 256, 0, stream>>>(states, carry, cumw);

  k_post<<<BT_ * H_, 64, 0, stream>>>(states, rbuf, kkbuf, vbuf, r_k, glin,
                                      ln_w, ln_b, xng);

  k_gemm_bf16<64, 128, 2, 4><<<gBig, 256, 0, stream>>>(xng, WoT, out, BT_, C_, C_);
}
